// LLMGRU4rec_36318243455037
// MI455X (gfx1250) — matrix operand order repaired and checked by reading, not compiled, not run on hardware
//
#include <hip/hip_runtime.h>
#include <hip/hip_bf16.h>
#include <math.h>

#define B_   128
#define T_   200
#define LLM_ 4096
#define H_   256
#define H3_  768
#define MBT  (B_*T_)      // 25600
#define M2   (2*MBT)      // 51200

typedef __attribute__((ext_vector_type(16))) __bf16 v16bf;
typedef __attribute__((ext_vector_type(8)))  float  v8f;
typedef __attribute__((ext_vector_type(4)))  unsigned int v4u;

union BFrag { unsigned int w[8]; v4u q[2]; v16bf v; };

__device__ __forceinline__ unsigned short f32_to_bf16(float f) {
  unsigned int u = __float_as_uint(f);
  u = (u + 0x7FFFu + ((u >> 16) & 1u)) >> 16;
  return (unsigned short)u;
}
// pack {hi16(lo), hi16(hi)} -> one dword with a single v_perm_b32 (truncating bf16)
__device__ __forceinline__ unsigned int permpack(float lo, float hi) {
  return __builtin_amdgcn_perm(__float_as_uint(hi), __float_as_uint(lo), 0x07060302u);
}

// ---- CDNA5 async global->LDS (ASYNCcnt) helpers, inline asm per ISA 15.18.3 ----
__device__ __forceinline__ void async_b128(unsigned int lds_off, const void* gaddr) {
  asm volatile("global_load_async_to_lds_b128 %0, %1, off"
               :: "v"(lds_off), "v"(gaddr) : "memory");
}
__device__ __forceinline__ void wait_async4() {
  asm volatile("s_wait_asynccnt 4" ::: "memory");
}
__device__ __forceinline__ void wait_async0() {
  asm volatile("s_wait_asynccnt 0" ::: "memory");
}

// A-matrix 16x32 bf16 (ISA 7.12.2): per lane two contiguous 8-elem chunks:
//   [k0+8*half, +8) -> w[0..3],  [k0+16+8*half, +8) -> w[4..7]
__device__ __forceinline__ v16bf fragA_bf16(const unsigned short* __restrict__ row, int k0, int half) {
  BFrag f;
  f.q[0] = *(const v4u*)(row + k0 + half * 8);
  f.q[1] = *(const v4u*)(row + k0 + 16 + half * 8);
  return f.v;
}
// B-matrix 32x16 bf16: per lane two 8-elem chunks at k0 + 8*half and k0 + 16 + 8*half
__device__ __forceinline__ v16bf fragB_bf16(const unsigned short* __restrict__ row, int k0, int half) {
  BFrag f;
  f.q[0] = *(const v4u*)(row + k0 + half * 8);
  f.q[1] = *(const v4u*)(row + k0 + 16 + half * 8);
  return f.v;
}
// raw fp32 A chunk (software pipelined) + pack with v_perm
struct ARaw { float4 p0, p1, p2, p3; };
__device__ __forceinline__ ARaw loadA_raw(const float* __restrict__ row, int k0, int half) {
  ARaw x;
  const float4* a = (const float4*)(row + k0 + half * 8);
  const float4* b = (const float4*)(row + k0 + 16 + half * 8);
  x.p0 = a[0]; x.p1 = a[1]; x.p2 = b[0]; x.p3 = b[1];
  return x;
}
__device__ __forceinline__ v16bf packA(const ARaw& x) {
  BFrag f;
  f.w[0] = permpack(x.p0.x, x.p0.y); f.w[1] = permpack(x.p0.z, x.p0.w);
  f.w[2] = permpack(x.p1.x, x.p1.y); f.w[3] = permpack(x.p1.z, x.p1.w);
  f.w[4] = permpack(x.p2.x, x.p2.y); f.w[5] = permpack(x.p2.z, x.p2.w);
  f.w[6] = permpack(x.p3.x, x.p3.y); f.w[7] = permpack(x.p3.z, x.p3.w);
  return f.v;
}
__device__ __forceinline__ v8f wmma_bf16(v16bf a, v16bf b, v8f c) {
  return __builtin_amdgcn_wmma_f32_16x16x32_bf16(false, a, false, b, (short)0, c, false, false);
}
__device__ __forceinline__ float geluf(float x) {
  return 0.5f * x * (1.0f + erff(x * 0.70710678118654752f));
}
__device__ __forceinline__ float sigmf(float x) {
  return 1.0f / (1.0f + __expf(-x));
}

// ---------------- weight fp32 -> bf16 conversion ----------------
__global__ __launch_bounds__(256) void k_cvt(const float* __restrict__ src,
                                             unsigned short* __restrict__ dst, int n) {
  int i = blockIdx.x * 256 + threadIdx.x;
  if (i < n) dst[i] = f32_to_bf16(src[i]);
}

// ---------------- Stage 1: Y1 = gelu([inter;neg] @ fc_W^T + fc_b), K=4096 ----------------
// wave = full 16x256 row-strip; B panel staged in LDS via async loads, double buffered
__global__ __launch_bounds__(256) void k_fc(const float* __restrict__ inter,
                                            const float* __restrict__ neg,
                                            const unsigned short* __restrict__ Wbf,
                                            const float* __restrict__ bias,
                                            unsigned short* __restrict__ y) {
  __shared__ __attribute__((aligned(32))) unsigned short Bp[2][256 * 32]; // 2 x 16KB
  const int tid  = threadIdx.x;
  const int lane = tid & 31;
  const int mt   = blockIdx.x * 8 + (tid >> 5);   // 0..3199
  const int half = lane >> 4, r = lane & 15;
  const int rowA = mt * 16 + r;
  const float* arow = (rowA < MBT) ? (inter + (size_t)rowA * LLM_)
                                   : (neg + (size_t)(rowA - MBT) * LLM_);
  // thread tid stages weight row n=tid: 64 bytes (4 x b128) per k-chunk
  const unsigned short* wsrc = Wbf + (size_t)tid * LLM_;
  unsigned int ldsrow[2];
  ldsrow[0] = (unsigned int)(size_t)(&Bp[0][tid * 32]);
  ldsrow[1] = (unsigned int)(size_t)(&Bp[1][tid * 32]);

  v8f acc[16];
#pragma unroll
  for (int j = 0; j < 16; ++j) acc[j] = 0.0f;

  // prologue: stage k-chunk 0 into buffer 0
  {
    const unsigned short* g = wsrc;
    async_b128(ldsrow[0] +  0, g);
    async_b128(ldsrow[0] + 16, g + 8);
    async_b128(ldsrow[0] + 32, g + 16);
    async_b128(ldsrow[0] + 48, g + 24);
  }
  ARaw araw = loadA_raw(arow, 0, half);

  for (int k0 = 0; k0 < LLM_; k0 += 32) {
    const int kn  = k0 + 32;
    const int cur = (k0 >> 5) & 1;
    ARaw anext;
    if (kn < LLM_) {
      // issue next B panel into the other buffer (consumed last iter)
      const unsigned short* g = wsrc + kn;
      const unsigned int lr = ldsrow[cur ^ 1];
      async_b128(lr +  0, g);
      async_b128(lr + 16, g + 8);
      async_b128(lr + 32, g + 16);
      async_b128(lr + 48, g + 24);
      anext = loadA_raw(arow, kn, half);
      wait_async4();            // group k0 complete (in-order), k0+32 in flight
    } else {
      wait_async0();
    }
    __syncthreads();            // all waves' group-k0 data visible in LDS
    v16bf a = packA(araw);
    const unsigned short* panel = &Bp[cur][0];
#pragma unroll
    for (int j = 0; j < 16; ++j) {
      v16bf b = fragB_bf16(panel + (j * 16 + r) * 32, 0, half);
      acc[j] = wmma_bf16(a, b, acc[j]);
    }
    __syncthreads();            // panel consumed; safe to overwrite next iter
    araw = anext;
  }
#pragma unroll
  for (int j = 0; j < 16; ++j) {
    const int n = j * 16 + r;
    const float bv = bias[n];
#pragma unroll
    for (int e = 0; e < 8; ++e) {
      const int m = mt * 16 + half * 8 + e;
      y[(size_t)m * H_ + n] = f32_to_bf16(geluf(acc[j][e] + bv));
    }
  }
}

// ---------------- Stage 2: Y2 = gelu(Y1 @ W1^T + b1), K=256, bf16 in/out ----------------
__global__ __launch_bounds__(256) void k_mlp(const unsigned short* __restrict__ X,
                                             const unsigned short* __restrict__ Wbf,
                                             const float* __restrict__ bias,
                                             unsigned short* __restrict__ y) {
  const int lane = threadIdx.x & 31;
  const int mt   = blockIdx.x * 8 + (threadIdx.x >> 5);
  const int half = lane >> 4, r = lane & 15;
  const unsigned short* arow = X + (size_t)(mt * 16 + r) * H_;
  v8f acc[16];
#pragma unroll
  for (int j = 0; j < 16; ++j) acc[j] = 0.0f;
  for (int k0 = 0; k0 < H_; k0 += 32) {
    v16bf a = fragA_bf16(arow, k0, half);
#pragma unroll
    for (int j = 0; j < 16; ++j) {
      v16bf b = fragB_bf16(Wbf + (size_t)(j * 16 + r) * H_, k0, half);
      acc[j] = wmma_bf16(a, b, acc[j]);
    }
  }
#pragma unroll
  for (int j = 0; j < 16; ++j) {
    const int n = j * 16 + r;
    const float bv = bias[n];
#pragma unroll
    for (int e = 0; e < 8; ++e) {
      const int m = mt * 16 + half * 8 + e;
      y[(size_t)m * H_ + n] = f32_to_bf16(geluf(acc[j][e] + bv));
    }
  }
}

// ---------------- Stage 3: embs = gelu(Y2 @ W2^T + b2); scatter outputs ----------------
__global__ __launch_bounds__(256) void k_mlp_out(const unsigned short* __restrict__ X,
                                                 const unsigned short* __restrict__ Wbf,
                                                 const float* __restrict__ bias,
                                                 unsigned short* __restrict__ embs_bf,
                                                 float* __restrict__ out2,
                                                 float* __restrict__ out3,
                                                 float* __restrict__ out4) {
  const int lane = threadIdx.x & 31;
  const int mt   = blockIdx.x * 8 + (threadIdx.x >> 5);
  const int half = lane >> 4, r = lane & 15;
  const unsigned short* arow = X + (size_t)(mt * 16 + r) * H_;
  v8f acc[16];
#pragma unroll
  for (int j = 0; j < 16; ++j) acc[j] = 0.0f;
  for (int k0 = 0; k0 < H_; k0 += 32) {
    v16bf a = fragA_bf16(arow, k0, half);
#pragma unroll
    for (int j = 0; j < 16; ++j) {
      v16bf b = fragB_bf16(Wbf + (size_t)(j * 16 + r) * H_, k0, half);
      acc[j] = wmma_bf16(a, b, acc[j]);
    }
  }
#pragma unroll
  for (int j = 0; j < 16; ++j) {
    const int n = j * 16 + r;
    const float bv = bias[n];
#pragma unroll
    for (int e = 0; e < 8; ++e) {
      const int m = mt * 16 + half * 8 + e;
      const float v = geluf(acc[j][e] + bv);
      if (m < MBT) {                       // interaction branch -> embs
        const int b = m / T_, t = m % T_;
        out4[((size_t)b * T_ + t) * (2 * H_) + n] = v;
        if (t >= 1) out2[((size_t)b * (T_ - 1) + (t - 1)) * H_ + n] = v;
        embs_bf[(size_t)m * H_ + n] = f32_to_bf16(v);
      } else {                             // neg branch
        const int mm = m - MBT;
        const int b = mm / T_, t = mm % T_;
        if (t < T_ - 1) out3[((size_t)b * (T_ - 1) + t) * H_ + n] = v;
      }
    }
  }
}

// ---------------- gi0 = embs @ Wih0^T + bih0  (all T precomputed) ----------------
__global__ __launch_bounds__(256) void k_gi0(const unsigned short* __restrict__ X,
                                             const unsigned short* __restrict__ Wbf,
                                             const float* __restrict__ bias,
                                             float* __restrict__ gi) {
  const int lane = threadIdx.x & 31;
  const int wid  = blockIdx.x * 8 + (threadIdx.x >> 5);
  const int mt = wid / 12, ns = wid % 12;
  const int half = lane >> 4, r = lane & 15;
  const unsigned short* arow = X + (size_t)(mt * 16 + r) * H_;
  const int nb = ns * 64;
  v8f acc[4];
#pragma unroll
  for (int j = 0; j < 4; ++j) acc[j] = 0.0f;
  for (int k0 = 0; k0 < H_; k0 += 32) {
    v16bf a = fragA_bf16(arow, k0, half);
#pragma unroll
    for (int j = 0; j < 4; ++j) {
      v16bf b = fragB_bf16(Wbf + (size_t)(nb + j * 16 + r) * H_, k0, half);
      acc[j] = wmma_bf16(a, b, acc[j]);
    }
  }
#pragma unroll
  for (int j = 0; j < 4; ++j) {
    const int n = nb + j * 16 + r;
    const float bv = bias[n];
#pragma unroll
    for (int e = 0; e < 8; ++e) {
      const int m = mt * 16 + half * 8 + e;
      gi[(size_t)m * H3_ + n] = acc[j][e] + bv;
    }
  }
}

// ---------------- GRU layer 0, one time step; gates fully in registers ----------------
__global__ __launch_bounds__(256) void k_gru0(int t,
                                              const float* __restrict__ gi0,
                                              const unsigned short* __restrict__ Whh0bf,
                                              const float* __restrict__ bhh0,
                                              const int* __restrict__ mask,
                                              float* __restrict__ h0,
                                              unsigned short* __restrict__ h0bf,
                                              unsigned short* __restrict__ h0nbf) {
  const int w = threadIdx.x >> 5, lane = threadIdx.x & 31;
  const int half = lane >> 4, r = lane & 15;
  const int rowA = blockIdx.x * 16 + r;
  const unsigned short* arow = h0bf + (size_t)rowA * H_;
  const int nb = w * 32;
  v8f ar[2], az[2], an[2];
#pragma unroll
  for (int j = 0; j < 2; ++j) { ar[j] = 0.0f; az[j] = 0.0f; an[j] = 0.0f; }
  for (int k0 = 0; k0 < H_; k0 += 32) {
    v16bf a = fragA_bf16(arow, k0, half);
#pragma unroll
    for (int j = 0; j < 2; ++j) {
      const int n = nb + j * 16 + r;
      ar[j] = wmma_bf16(a, fragB_bf16(Whh0bf + (size_t)(n)       * H_, k0, half), ar[j]);
      az[j] = wmma_bf16(a, fragB_bf16(Whh0bf + (size_t)(256 + n) * H_, k0, half), az[j]);
      an[j] = wmma_bf16(a, fragB_bf16(Whh0bf + (size_t)(512 + n) * H_, k0, half), an[j]);
    }
  }
  __syncthreads();   // all waves done reading h0bf before anyone overwrites it
#pragma unroll
  for (int j = 0; j < 2; ++j) {
    const int n = nb + j * 16 + r;
    const float br = bhh0[n], bz = bhh0[256 + n], bn = bhh0[512 + n];
#pragma unroll
    for (int e = 0; e < 8; ++e) {
      const int brow = blockIdx.x * 16 + half * 8 + e;
      const size_t gb = ((size_t)brow * T_ + t) * H3_;
      const float rr = sigmf(gi0[gb + n] + ar[j][e] + br);
      const float zz = sigmf(gi0[gb + 256 + n] + az[j][e] + bz);
      const float nn = tanhf(gi0[gb + 512 + n] + rr * (an[j][e] + bn));
      const float hp = h0[(size_t)brow * H_ + n];
      const float hn = (1.0f - zz) * nn + zz * hp;
      h0nbf[(size_t)brow * H_ + n] = f32_to_bf16(hn);
      const float mv = mask[brow * T_ + t] ? 1.0f : 0.0f;
      const float hm = hn * mv;
      h0[(size_t)brow * H_ + n] = hm;
      h0bf[(size_t)brow * H_ + n] = f32_to_bf16(hm);
    }
  }
}

// ---------------- GRU layer 1 + output head, one time step ----------------
__global__ __launch_bounds__(256) void k_gru1(int t,
                                              const unsigned short* __restrict__ Wih1bf,
                                              const unsigned short* __restrict__ Whh1bf,
                                              const float* __restrict__ bih1,
                                              const float* __restrict__ bhh1,
                                              const unsigned short* __restrict__ h2oWbf,
                                              const float* __restrict__ h2ob,
                                              const int* __restrict__ mask,
                                              const unsigned short* __restrict__ h0nbf,
                                              float* __restrict__ h1,
                                              unsigned short* __restrict__ h1bf,
                                              float* __restrict__ out1,
                                              float* __restrict__ out4) {
  __shared__ __attribute__((aligned(32))) unsigned short h1n_lds[16 * H_];
  const int w = threadIdx.x >> 5, lane = threadIdx.x & 31;
  const int half = lane >> 4, r = lane & 15;
  const int rowA = blockIdx.x * 16 + r;
  const unsigned short* a0row = h0nbf + (size_t)rowA * H_;
  const unsigned short* a1row = h1bf + (size_t)rowA * H_;
  const int nb = w * 32;
  v8f acr[2], acz[2], agn[2], ahn[2];
#pragma unroll
  for (int j = 0; j < 2; ++j) { acr[j] = 0.0f; acz[j] = 0.0f; agn[j] = 0.0f; ahn[j] = 0.0f; }
  for (int k0 = 0; k0 < H_; k0 += 32) {
    v16bf a0 = fragA_bf16(a0row, k0, half);
    v16bf a1 = fragA_bf16(a1row, k0, half);
#pragma unroll
    for (int j = 0; j < 2; ++j) {
      const int n = nb + j * 16 + r;
      acr[j] = wmma_bf16(a0, fragB_bf16(Wih1bf + (size_t)(n)       * H_, k0, half), acr[j]);
      acr[j] = wmma_bf16(a1, fragB_bf16(Whh1bf + (size_t)(n)       * H_, k0, half), acr[j]);
      acz[j] = wmma_bf16(a0, fragB_bf16(Wih1bf + (size_t)(256 + n) * H_, k0, half), acz[j]);
      acz[j] = wmma_bf16(a1, fragB_bf16(Whh1bf + (size_t)(256 + n) * H_, k0, half), acz[j]);
      agn[j] = wmma_bf16(a0, fragB_bf16(Wih1bf + (size_t)(512 + n) * H_, k0, half), agn[j]);
      ahn[j] = wmma_bf16(a1, fragB_bf16(Whh1bf + (size_t)(512 + n) * H_, k0, half), ahn[j]);
    }
  }
  __syncthreads();   // h1bf fully consumed before overwrite
#pragma unroll
  for (int j = 0; j < 2; ++j) {
    const int n = nb + j * 16 + r;
    const float br = bih1[n] + bhh1[n];
    const float bz = bih1[256 + n] + bhh1[256 + n];
    const float bgn = bih1[512 + n];
    const float bhn = bhh1[512 + n];
#pragma unroll
    for (int e = 0; e < 8; ++e) {
      const int mloc = half * 8 + e;
      const int brow = blockIdx.x * 16 + mloc;
      const float rr = sigmf(acr[j][e] + br);
      const float zz = sigmf(acz[j][e] + bz);
      const float nn = tanhf(agn[j][e] + bgn + rr * (ahn[j][e] + bhn));
      const float hp = h1[(size_t)brow * H_ + n];
      const float hn = (1.0f - zz) * nn + zz * hp;
      h1n_lds[mloc * H_ + n] = f32_to_bf16(hn);
      const float mv = mask[brow * T_ + t] ? 1.0f : 0.0f;
      const float hm = hn * mv;
      h1[(size_t)brow * H_ + n] = hm;
      h1bf[(size_t)brow * H_ + n] = f32_to_bf16(hm);
    }
  }
  __syncthreads();   // h1n tile staged in LDS
  // logit = tanh(h1n @ h2o_W^T + h2o_b): wave handles 2 N-tiles
  v8f lacc[2];
#pragma unroll
  for (int j = 0; j < 2; ++j) lacc[j] = 0.0f;
  const unsigned short* arow = h1n_lds + (size_t)r * H_;
  for (int k0 = 0; k0 < H_; k0 += 32) {
    v16bf a = fragA_bf16(arow, k0, half);
#pragma unroll
    for (int j = 0; j < 2; ++j) {
      const int n = (w * 2 + j) * 16 + r;
      lacc[j] = wmma_bf16(a, fragB_bf16(h2oWbf + (size_t)n * H_, k0, half), lacc[j]);
    }
  }
#pragma unroll
  for (int j = 0; j < 2; ++j) {
    const int n = (w * 2 + j) * 16 + r;
    const float bv = h2ob[n];
#pragma unroll
    for (int e = 0; e < 8; ++e) {
      const int brow = blockIdx.x * 16 + half * 8 + e;
      const float v = tanhf(lacc[j][e] + bv);
      out4[((size_t)brow * T_ + t) * (2 * H_) + H_ + n] = v;
      if (t < T_ - 1) out1[((size_t)brow * (T_ - 1) + t) * H_ + n] = v;
    }
  }
}

extern "C" void kernel_launch(void* const* d_in, const int* in_sizes, int n_in,
                              void* d_out, int out_size, void* d_ws, size_t ws_size,
                              hipStream_t stream) {
  (void)in_sizes; (void)n_in; (void)out_size;
  const float* inter = (const float*)d_in[0];
  const int*   imask = (const int*)d_in[1];
  const float* neg   = (const float*)d_in[2];
  const float* fcW   = (const float*)d_in[3];
  const float* fcb   = (const float*)d_in[4];
  const float* W1    = (const float*)d_in[5];
  const float* b1    = (const float*)d_in[6];
  const float* W2    = (const float*)d_in[7];
  const float* b2    = (const float*)d_in[8];
  const float* Wih0  = (const float*)d_in[9];
  const float* Whh0  = (const float*)d_in[10];
  const float* bih0  = (const float*)d_in[11];
  const float* bhh0  = (const float*)d_in[12];
  const float* Wih1  = (const float*)d_in[13];
  const float* Whh1  = (const float*)d_in[14];
  const float* bih1  = (const float*)d_in[15];
  const float* bhh1  = (const float*)d_in[16];
  const float* h2oW  = (const float*)d_in[17];
  const float* h2ob  = (const float*)d_in[18];

  char* ws = (char*)d_ws;
  size_t o = 0;
  unsigned short* y1     = (unsigned short*)(ws + o); o += (size_t)M2 * H_ * 2;
  unsigned short* y2     = (unsigned short*)(ws + o); o += (size_t)M2 * H_ * 2;
  unsigned short* embs   = (unsigned short*)(ws + o); o += (size_t)MBT * H_ * 2;
  float* gi0             = (float*)(ws + o);          o += (size_t)MBT * H3_ * 4;
  unsigned short* fcWbf  = (unsigned short*)(ws + o); o += (size_t)H_ * LLM_ * 2;
  unsigned short* W1bf   = (unsigned short*)(ws + o); o += (size_t)H_ * H_ * 2;
  unsigned short* W2bf   = (unsigned short*)(ws + o); o += (size_t)H_ * H_ * 2;
  unsigned short* Wih0bf = (unsigned short*)(ws + o); o += (size_t)H3_ * H_ * 2;
  unsigned short* Whh0bf = (unsigned short*)(ws + o); o += (size_t)H3_ * H_ * 2;
  unsigned short* Wih1bf = (unsigned short*)(ws + o); o += (size_t)H3_ * H_ * 2;
  unsigned short* Whh1bf = (unsigned short*)(ws + o); o += (size_t)H3_ * H_ * 2;
  unsigned short* h2oWbf = (unsigned short*)(ws + o); o += (size_t)H_ * H_ * 2;
  float* h0              = (float*)(ws + o);          o += (size_t)B_ * H_ * 4;
  float* h1              = (float*)(ws + o);          o += (size_t)B_ * H_ * 4;
  unsigned short* h0bf   = (unsigned short*)(ws + o); o += (size_t)B_ * H_ * 2;
  unsigned short* h1bf   = (unsigned short*)(ws + o); o += (size_t)B_ * H_ * 2;
  unsigned short* h0nbf  = (unsigned short*)(ws + o); o += (size_t)B_ * H_ * 2;
  if (o > ws_size) return;  // workspace too small; nothing safe to do

  float* out1 = (float*)d_out;                       // final_output[:, :-1, :]
  float* out2 = out1 + (size_t)B_ * (T_ - 1) * H_;   // target_pos_embs
  float* out3 = out2 + (size_t)B_ * (T_ - 1) * H_;   // target_neg_embs
  float* out4 = out3 + (size_t)B_ * (T_ - 1) * H_;   // concat(embs, final_output)

  // zero recurrent state (f32 h0,h1 + bf16 h0bf,h1bf,h0nbf; contiguous region)
  hipMemsetAsync(h0, 0, (size_t)B_ * H_ * (4 + 4 + 2 + 2 + 2), stream);

  // weights -> bf16 (RNE), once per call
  k_cvt<<<(H_ * LLM_) / 256, 256, 0, stream>>>(fcW, fcWbf, H_ * LLM_);
  k_cvt<<<(H_ * H_) / 256, 256, 0, stream>>>(W1, W1bf, H_ * H_);
  k_cvt<<<(H_ * H_) / 256, 256, 0, stream>>>(W2, W2bf, H_ * H_);
  k_cvt<<<(H3_ * H_) / 256, 256, 0, stream>>>(Wih0, Wih0bf, H3_ * H_);
  k_cvt<<<(H3_ * H_) / 256, 256, 0, stream>>>(Whh0, Whh0bf, H3_ * H_);
  k_cvt<<<(H3_ * H_) / 256, 256, 0, stream>>>(Wih1, Wih1bf, H3_ * H_);
  k_cvt<<<(H3_ * H_) / 256, 256, 0, stream>>>(Whh1, Whh1bf, H3_ * H_);
  k_cvt<<<(H_ * H_) / 256, 256, 0, stream>>>(h2oW, h2oWbf, H_ * H_);

  // MLP chain over [interaction; neg] rows (M = 51200); waves own full 16x256 strips
  k_fc     <<<400, 256, 0, stream>>>(inter, neg, fcWbf, fcb, y1);
  k_mlp    <<<400, 256, 0, stream>>>(y1, W1bf, b1, y2);
  k_mlp_out<<<400, 256, 0, stream>>>(y2, W2bf, b2, embs, out2, out3, out4);

  // gi0 for all time steps
  k_gi0<<<2400, 256, 0, stream>>>(embs, Wih0bf, bih0, gi0);

  // sequential GRU scan
  for (int t = 0; t < T_; ++t) {
    k_gru0<<<8, 256, 0, stream>>>(t, gi0, Whh0bf, bhh0, imask, h0, h0bf, h0nbf);
    k_gru1<<<8, 256, 0, stream>>>(t, Wih1bf, Whh1bf, bih1, bhh1, h2oWbf, h2ob, imask,
                                  h0nbf, h1, h1bf, out1, out4);
  }
}
